// Reconstructeur_66812511256934
// MI455X (gfx1250) — compile-verified, matrix operand order checked by reading, not run on hardware
//
#include <hip/hip_runtime.h>
#include <cmath>

typedef __bf16 bf16_t;
typedef __attribute__((ext_vector_type(16))) __bf16 v16bf;
typedef __attribute__((ext_vector_type(8)))  float  v8f;
typedef __attribute__((ext_vector_type(4)))  unsigned int u32x4;

// ---------------- problem constants (fixed by setup_inputs) ----------------
constexpr int M_MLP = 16;
constexpr int LAT   = 128;          // latent size (broadcast to every point)
constexpr int DIN   = 130;          // latent + 2 coords
constexpr int N1    = 512, N2 = 256, N3 = 128;
constexpr int GRD   = 128;          // n_grid_step
constexpr int P     = GRD * GRD;    // 16384 points
constexpr int PT    = 64;           // points per workgroup
constexpr int PTT   = PT / 16;      // 4 p-subtiles (register-blocked per wave)
constexpr int WAVES = 8;
constexpr int TPB   = WAVES * 32;   // 256 threads (wave32)

// bf16 swizzled-weight regions inside d_ws (element offsets)
// layer-1 only keeps the coordinate columns k=128..129 (one K=32 fragment);
// the x-dependent part is folded into a precomputed bias c1.
constexpr size_t F1 = (size_t)M_MLP * (N1/16) * 1        * 512;  //   262,144
constexpr size_t F2 = (size_t)M_MLP * (N2/16) * (N1/32)  * 512;  // 2,097,152
constexpr size_t F3 = (size_t)M_MLP * (N3/16) * (N2/32)  * 512;  //   524,288
constexpr size_t F4 = (size_t)M_MLP * 1       * (N3/32)  * 512;  //    32,768
constexpr size_t O1 = 0, O2 = O1 + F1, O3 = O2 + F2, O4 = O3 + F3;
constexpr size_t WS_BF16_END = O4 + F4;                 // 2,916,352 elems
constexpr size_t C1_BYTE_OFF = WS_BF16_END * 2;         // 16B aligned
// total ws: ~5.83 MB weights + 32 KB c1

// -------- prologue A: fold latent into per-(m,n) bias: c1 = b1 + W1[:, :128].x
__global__ void latent_bias(const float* __restrict__ x, const float* __restrict__ W1,
                            const float* __restrict__ b1, float* __restrict__ c1)
{
    int i = blockIdx.x * blockDim.x + threadIdx.x;    // i = m*N1 + n
    if (i >= M_MLP * N1) return;
    const float* w = W1 + (size_t)i * DIN;
    float acc = b1[i];
#pragma unroll 4
    for (int k = 0; k < LAT; ++k) acc = fmaf(x[k], w[k], acc);
    c1[i] = acc;
}

// -------- prologue B: fp32 [m][N][K] weights -> bf16 WMMA B-fragments ------
// fragment layout: elem = ((((m*nTiles + nb)*kFrags + kb)*32 + lane)*16 + e)
// lane holds column n = nb*16 + (lane&15); element e -> K via the 16-bit
// operand interleave (ISA 7.12.2): K = ((e>>3)<<4) + (lane>>4)*8 + (e&7)
__global__ void swizzle_w(const float* __restrict__ W, bf16_t* __restrict__ dst,
                          int Nreal, int Kreal, int kOff,
                          int nTiles, int kFrags, int total)
{
    int i = blockIdx.x * blockDim.x + threadIdx.x;
    if (i >= total) return;
    const int e    = i & 15;
    const int lane = (i >> 4) & 31;
    const int rest = i >> 9;
    const int kb   = rest % kFrags;
    const int r2   = rest / kFrags;
    const int nb   = r2 % nTiles;
    const int m    = r2 / nTiles;
    const int n    = nb * 16 + (lane & 15);
    const int k    = kOff + kb * 32 + ((e >> 3) << 4) + (lane >> 4) * 8 + (e & 7);
    float v = 0.0f;
    if (n < Nreal && k < Kreal) v = W[((size_t)m * Nreal + n) * Kreal + k];
    dst[i] = (bf16_t)v;
}

// ------------------------- fragment load helpers ---------------------------
__device__ __forceinline__ v16bf load_a_frag(const bf16_t* in, int ldk,
                                             int prow0, int k0, int lane)
{
    const int row = lane & 15;
    const int hf  = lane >> 4;
    const bf16_t* p = in + (size_t)(prow0 + row) * ldk + k0 + hf * 8;
    union { v16bf v; u32x4 q[2]; } f;
    f.q[0] = *(const u32x4*)p;         // K = k0 + hf*8 + [0..7]
    f.q[1] = *(const u32x4*)(p + 16);  // K = k0 + 16 + hf*8 + [0..7]
    return f.v;
}

__device__ __forceinline__ v16bf load_b_frag(const bf16_t* w, int lane)
{
    union { v16bf v; u32x4 q[2]; } f;
    const u32x4* p = (const u32x4*)(w + (size_t)lane * 16);
    f.q[0] = p[0];
    f.q[1] = p[1];
    return f.v;
}

// fused GEMM+bias+ReLU layer, register-blocked: each wave owns column-tile
// nt and all 4 p-subtiles, so every B fragment is loaded once per 4 WMMAs.
template <int LDK_IN, int N_OUT, int KFRAGS>
__device__ __forceinline__ void layer_relu(const bf16_t* in, bf16_t* outLds,
                                           const bf16_t* wfragM,
                                           const float* __restrict__ biasM,
                                           int lane, int wave)
{
    constexpr int NT = N_OUT / 16;
    const int hf  = lane >> 4;
    const int col = lane & 15;
    for (int nt = wave; nt < NT; nt += WAVES) {
        const float cb = biasM[nt * 16 + col];
        v8f c[PTT];
#pragma unroll
        for (int p = 0; p < PTT; ++p)
#pragma unroll
            for (int v = 0; v < 8; ++v) c[p][v] = cb;
        const bf16_t* wf = wfragM + (size_t)nt * KFRAGS * 512;
#pragma unroll
        for (int kb = 0; kb < KFRAGS; ++kb) {
            v16bf b = load_b_frag(wf + kb * 512, lane);
#pragma unroll
            for (int p = 0; p < PTT; ++p) {
                v16bf a = load_a_frag(in, LDK_IN, p * 16, kb * 32, lane);
                c[p] = __builtin_amdgcn_wmma_f32_16x16x32_bf16(
                           false, a, false, b, (short)0, c[p], false, false);
            }
        }
        // D layout: lane = column, VGPR v = row (v + 8*(lane>>4))
#pragma unroll
        for (int p = 0; p < PTT; ++p) {
            bf16_t* o = outLds + (size_t)(p * 16 + 8 * hf) * N_OUT + nt * 16 + col;
#pragma unroll
            for (int v = 0; v < 8; ++v)
                o[(size_t)v * N_OUT] = (bf16_t)fmaxf(c[p][v], 0.0f);
        }
    }
}

// ------------------------------ main kernel --------------------------------
__global__ __launch_bounds__(TPB) void mlp_fused(
    const float* __restrict__ c1, const float* __restrict__ b2,
    const float* __restrict__ b3, const float* __restrict__ b4,
    const bf16_t* __restrict__ w1s, const bf16_t* __restrict__ w2s,
    const bf16_t* __restrict__ w3s, const bf16_t* __restrict__ w4s,
    float* __restrict__ out)
{
    __shared__ __align__(16) bf16_t bufE[PT * N2]; // coords (PT x 32) -> h2 (PT x 256)
    __shared__ __align__(16) bf16_t bufO[PT * N1]; // h1 (PT x 512)    -> h3 (PT x 128)

    const int tid   = threadIdx.x;
    const int lane  = tid & 31;
    const int wave  = tid >> 5;
    const int m     = blockIdx.y;
    const int pBase = blockIdx.x * PT;

    // stage coordinate tile: columns 0/1 = meshgrid X/Y (k=128/129), rest 0
    for (int i = tid; i < PT * 32; i += TPB) {
        const int p = i >> 5;
        const int k = i & 31;
        float v = 0.0f;
        if (k < 2) {
            const int gp = pBase + p;
            const int cc = gp & (GRD - 1);  // X varies fastest (meshgrid 'xy')
            const int rr = gp >> 7;
            const float t = (k == 0) ? (float)cc : (float)rr;
            v = -1.0f + (2.0f / (float)(GRD - 1)) * t;
        }
        bufE[i] = (bf16_t)v;
    }
    __syncthreads();

    layer_relu<32, N1, 1>(bufE, bufO,
        w1s + (size_t)m * (N1/16) * 512, c1 + m * N1, lane, wave);
    __syncthreads();
    layer_relu<N1, N2, N1/32>(bufO, bufE,
        w2s + (size_t)m * (N2/16) * (N1/32) * 512, b2 + m * N2, lane, wave);
    __syncthreads();
    layer_relu<N2, N3, N2/32>(bufE, bufO,
        w3s + (size_t)m * (N3/16) * (N2/32) * 512, b3 + m * N3, lane, wave);
    __syncthreads();

    // layer 4: [PT,128] x [128, 3(->16 padded)] -> tanh -> global out
    {
        const bf16_t* wf4 = w4s + (size_t)m * (N3/32) * 512;
        const int hf = lane >> 4, col = lane & 15;
        for (int pt = wave; pt < PTT; pt += WAVES) {
            const float cb = (col < 3) ? b4[m * 3 + col] : 0.0f;
            v8f c;
#pragma unroll
            for (int v = 0; v < 8; ++v) c[v] = cb;
#pragma unroll
            for (int kb = 0; kb < N3 / 32; ++kb) {
                v16bf a = load_a_frag(bufO, N3, pt * 16, kb * 32, lane);
                v16bf b = load_b_frag(wf4 + kb * 512, lane);
                c = __builtin_amdgcn_wmma_f32_16x16x32_bf16(
                        false, a, false, b, (short)0, c, false, false);
            }
            if (col < 3) {
#pragma unroll
                for (int v = 0; v < 8; ++v) {
                    const int gp = pBase + pt * 16 + v + 8 * hf;
                    out[((size_t)m * P + gp) * 3 + col] = tanhf(c[v]);
                }
            }
        }
    }
}

// ------------------------------- launcher ----------------------------------
extern "C" void kernel_launch(void* const* d_in, const int* in_sizes, int n_in,
                              void* d_out, int out_size, void* d_ws, size_t ws_size,
                              hipStream_t stream)
{
    const float* x  = (const float*)d_in[0];
    const float* W1 = (const float*)d_in[1];
    const float* b1 = (const float*)d_in[2];
    const float* W2 = (const float*)d_in[3];
    const float* b2 = (const float*)d_in[4];
    const float* W3 = (const float*)d_in[5];
    const float* b3 = (const float*)d_in[6];
    const float* W4 = (const float*)d_in[7];
    const float* b4 = (const float*)d_in[8];
    bf16_t* ws = (bf16_t*)d_ws;
    float*  c1 = (float*)((char*)d_ws + C1_BYTE_OFF);

    latent_bias<<<(M_MLP * N1 + 255) / 256, 256, 0, stream>>>(x, W1, b1, c1);

    struct SwArg { const float* W; bf16_t* dst; int N, K, kOff, nT, kF; };
    const SwArg sw[4] = {
        { W1, ws + O1, N1, DIN, LAT, N1/16, 1     },  // only coord columns
        { W2, ws + O2, N2, N1,  0,   N2/16, N1/32 },
        { W3, ws + O3, N3, N2,  0,   N3/16, N2/32 },
        { W4, ws + O4, 3,  N3,  0,   1,     N3/32 },
    };
    for (int i = 0; i < 4; ++i) {
        const int total = M_MLP * sw[i].nT * sw[i].kF * 512;
        swizzle_w<<<(total + 255) / 256, 256, 0, stream>>>(
            sw[i].W, sw[i].dst, sw[i].N, sw[i].K, sw[i].kOff, sw[i].nT, sw[i].kF, total);
    }

    mlp_fused<<<dim3(P / PT, M_MLP), TPB, 0, stream>>>(
        c1, b2, b3, b4, ws + O1, ws + O2, ws + O3, ws + O4, (float*)d_out);
}
